// TransformerRegressor_2525440770259
// MI455X (gfx1250) — matrix operand order repaired and checked by reading, not compiled, not run on hardware
//
#include <hip/hip_runtime.h>
#include <cstdint>
#include <math.h>

// CDNA5 (gfx1250) megakernel for the KAN/Mamba/attention regressor.
// wave32, WMMA f32_16x16x32_f16, double-buffered TDM weight staging
// (overlaps DMA with spline expansion + WMMA), async-to-LDS input staging.
// Weights are pre-swizzled into WMMA B-fragment tile order so every LDS
// fragment read is a contiguous b128 load.

typedef __attribute__((ext_vector_type(16))) _Float16 v16h;
typedef __attribute__((ext_vector_type(8)))  float    v8f;
typedef __attribute__((ext_vector_type(4)))  unsigned int tdm4u;
typedef __attribute__((ext_vector_type(8)))  int          tdm8i;
typedef __attribute__((ext_vector_type(4)))  int          tdm4i;

#define THREADS   128          // 4 waves
#define TILE_M    16           // batch rows per workgroup (WMMA M)
#define STRIDE    128          // tile row stride (elements)
#define SW_HALFS  4096         // one LDS weight slab (halves) -- double buffered
#define BATCH     1024

// ---------------------------------------------------------------- helpers ---

__device__ __forceinline__ unsigned lds_off(const void* p) {
  // ISA 10.2: for LDS-aperture flat addresses, the LDS offset is the low 32 bits of the address.
  return (unsigned)(unsigned long long)(uintptr_t)p;
}

__device__ __forceinline__ void async_copy_b32(unsigned lds_byte,
                                               unsigned long long gbase,
                                               unsigned gbyte) {
  asm volatile("global_load_async_to_lds_b32 %0, %1, %2"
               :: "v"(lds_byte), "v"(gbyte), "s"(gbase) : "memory");
}

__device__ __forceinline__ void wait_async0() {
  asm volatile("s_wait_asynccnt 0x0" ::: "memory");
}

// One TDM descriptor: contiguous 1-row tile of `bytes` (4-byte elements).
__device__ __forceinline__ void tdm_load_slab(unsigned lds_byte,
                                              const void* gsrc,
                                              unsigned bytes) {
  unsigned long long ga = (unsigned long long)(uintptr_t)gsrc;
  unsigned words = bytes >> 2;                    // 4-byte elements
  tdm4u g0;
  g0[0] = 1u;                                     // count=1, user mode
  g0[1] = lds_byte;                               // lds_addr [63:32]
  g0[2] = (unsigned)ga;                           // global_addr low
  g0[3] = ((unsigned)(ga >> 32) & 0x01ffffffu)    // global_addr [120:96]
          | 0x80000000u;                          // type=2 -> bits[127:126]=10b
  tdm8i g1;
  g1[0] = (int)(2u << 16);                        // data_size=2 (4B); mask=0
  g1[1] = (int)((words & 0xffffu) << 16);         // tensor_dim0 low16 @bits[63:48]
  g1[2] = (int)((words >> 16) | (1u << 16));      // dim0 hi16 | tensor_dim1=1
  g1[3] = (int)((words & 0xffffu) << 16);         // tile_dim0 @bits[127:112]
  g1[4] = 1;                                      // tile_dim1=1
  g1[5] = (int)words;                             // tensor_dim0_stride low32
  g1[6] = 0;
  g1[7] = 0;
  tdm4i z4 = {0, 0, 0, 0};
#if defined(__clang_major__) && (__clang_major__ >= 23)
  tdm8i z8 = {0, 0, 0, 0, 0, 0, 0, 0};
  __builtin_amdgcn_tensor_load_to_lds(g0, g1, z4, z4, z8, 0);
#else
  __builtin_amdgcn_tensor_load_to_lds(g0, g1, z4, z4, 0);
#endif
}

// ------------------------------------------------ fused KAN dense (WMMA) ----
// input : sTile[16 x fi] (f32).  output: sTile[16 x fo] (f32).
// K = 9*fi  ( silu(x) [fi]  ||  8 cubic B-spline bases per input [8*fi] ).
// Weights arrive pre-swizzled as 32x16 tiles of 512 halves:
//   tile = ktile*(fo/16)+nb ; within tile: lane*16+e, k=ktile*32+(lane>>4)*16+e,
//   n = nb*16+(lane&15)  -> B fragment is one contiguous v16h per lane.
// TDM pipeline: slab 0 issued before the expansion (overlap DMA with VALU);
// inside the loop wave0 issues slab s+1 then waits TENSORcnt<=1 (in-order
// completion guarantees slab s is resident).
__device__ const _Float16* kan_gemm(const _Float16* wp, int fi, int fo,
                                    float* sTile, _Float16* sAct, _Float16* sW,
                                    int tid) {
  const int lda = fi * 9;
  const int K = lda;                          // multiple of 32 (fi in {32,64,128})
  const int wave = tid >> 5, lane = tid & 31;
  const int m = lane & 15, hi = lane >> 4;
  const int slabK = SW_HALFS / fo;            // 128 / 64 / 32 K-rows per slab
  const int nslab = (K + slabK - 1) / slabK;
  const int ntiles = fo >> 4;

  // prologue: start DMA of slab 0 so it overlaps the spline expansion
  if (wave == 0) {
    int c0 = (K < slabK) ? K : slabK;
    tdm_load_slab(lds_off(sW), wp, (unsigned)(c0 * fo) * 2u);
  }

  // ---- expansion: silu + Cox-de-Boor bases (uniform grid, 12 knots) ----
  for (int i = tid; i < TILE_M * fi; i += THREADS) {
    int r = i / fi, f = i % fi;
    float xv = sTile[r * STRIDE + f];
    sAct[r * lda + f] = (_Float16)(xv / (1.f + __expf(-xv)));
    float bases[11];
#pragma unroll
    for (int t = 0; t < 11; ++t) {
      float g0 = -1.f + (float)(t - 3) * 0.4f;
      bases[t] = (xv >= g0 && xv < g0 + 0.4f) ? 1.f : 0.f;
    }
#pragma unroll
    for (int k = 1; k <= 3; ++k)
#pragma unroll
      for (int t = 0; t < 10; ++t) {
        if (t < 11 - k) {
          float gi  = -1.f + (float)(t - 3) * 0.4f;
          float gik = gi + 0.4f * (float)k;
          bases[t] = (xv - gi) / (gik - gi) * bases[t] +
                     (gik + 0.4f - xv) / (0.4f * (float)k) * bases[t + 1];
        }
      }
#pragma unroll
    for (int j = 0; j < 8; ++j)
      sAct[r * lda + fi + f * 8 + j] = (_Float16)bases[j];
  }
  __syncthreads();

  v8f acc0 = {0,0,0,0,0,0,0,0}, acc1 = {0,0,0,0,0,0,0,0};

  for (int s = 0; s < nslab; ++s) {
    const int k0  = s * slabK;
    const int cur = (K - k0 < slabK) ? (K - k0) : slabK;
    const _Float16* buf = sW + ((s & 1) ? SW_HALFS : 0);

    if (wave == 0) {                          // wave-uniform producer
      if (s + 1 < nslab) {
        int nk0  = (s + 1) * slabK;
        int ncur = (K - nk0 < slabK) ? (K - nk0) : slabK;
        __builtin_prefetch(wp + (size_t)(nk0 + slabK) * fo, 0, 1);
        tdm_load_slab(lds_off(sW + (((s + 1) & 1) ? SW_HALFS : 0)),
                      wp + (size_t)nk0 * fo, (unsigned)(ncur * fo) * 2u);
        __builtin_amdgcn_s_wait_tensorcnt(1); // slab s done (in-order TDM)
      } else {
        __builtin_amdgcn_s_wait_tensorcnt(0);
      }
    }
    __syncthreads();                          // slab s visible to all waves

    for (int kk = 0; kk < cur; kk += 32) {
      v16h a;                                 // 16-bit A 16x32 per-lane layout
#pragma unroll
      for (int e = 0; e < 8; ++e) {
        a[e]     = sAct[m * lda + k0 + kk + hi * 16 + e];
        a[8 + e] = sAct[m * lda + k0 + kk + hi * 16 + 8 + e];
      }
#pragma unroll
      for (int t = 0; t < 2; ++t) {
        int nb = wave + 4 * t;                // wave-uniform: EXEC stays all-1s
        if (nb < ntiles) {
          const _Float16* bp = buf + ((size_t)((kk >> 5) * ntiles + nb) << 9)
                                   + (lane << 4);
          v16h b = *(const v16h*)bp;          // contiguous: 2x ds_load_b128
          v8f c = t ? acc1 : acc0;
          c = __builtin_amdgcn_wmma_f32_16x16x32_f16(false, a, false, b,
                                                     (short)0, c, false, false);
          if (t) acc1 = c; else acc0 = c;
        }
      }
    }
    __syncthreads();                          // done reading; slab may be reused
  }
#pragma unroll
  for (int t = 0; t < 2; ++t) {
    int nb = wave + 4 * t;
    if (nb < ntiles) {
      v8f c = t ? acc1 : acc0;
#pragma unroll
      for (int v = 0; v < 8; ++v)             // C/D layout: VGPR v, M = v + 8*hi
        sTile[(v + 8 * hi) * STRIDE + nb * 16 + m] = c[v];
    }
  }
  __syncthreads();
  return wp + (size_t)K * fo;
}

// -------------------------------------------------------------- epilogues ---
// mode 0: silu   mode 1: gelu(exact)   mode 2: residual add + RMSNorm
// residual kept in f16 to fit the 64KB LDS arena with double-buffered weights.
__device__ void epilogue(float* sTile, _Float16* sRes, float* sRms, int fo,
                         int tid, int mode, bool save) {
  if (mode == 2) {
    if (tid < TILE_M) {
      float ss = 0.f;
      for (int f = 0; f < fo; ++f) {
        float v = sTile[tid * STRIDE + f] + (float)sRes[tid * STRIDE + f];
        ss += v * v;
      }
      sRms[tid] = rsqrtf(ss / (float)fo + 1.1920929e-07f);
    }
    __syncthreads();
    for (int i = tid; i < TILE_M * fo; i += THREADS) {
      int r = i / fo, f = i % fo;
      float v = (sTile[r * STRIDE + f] + (float)sRes[r * STRIDE + f]) * sRms[r];
      sTile[r * STRIDE + f] = v;
      sRes[r * STRIDE + f]  = (_Float16)v;
    }
  } else {
    for (int i = tid; i < TILE_M * fo; i += THREADS) {
      int r = i / fo, f = i % fo;
      float v = sTile[r * STRIDE + f];
      v = (mode == 0) ? v / (1.f + __expf(-v))
                      : 0.5f * v * (1.f + erff(v * 0.70710678f));
      sTile[r * STRIDE + f] = v;
      if (save) sRes[r * STRIDE + f] = (_Float16)v;
    }
  }
  __syncthreads();
}

// ------------------------------------------------------------ main kernel ---
// LDS arena (64 KB):
//   [     0, 36864)  sAct   16 x 1152 f16      (K_max = 9*128)
//   [ 36864, 53248)  sW     2 x 4096 f16 slabs (TDM double buffer)
//   [ 53248, 61440)  sTile  16 x 128 f32
//   [ 61440, 65536)  sRes   16 x 128 f16
//   sRaw (16x20 f32) and sRms (16 f32) alias sAct (disjoint lifetimes).
__global__ __launch_bounds__(THREADS)
void fused_regressor(const float* __restrict__ x,      // [B,1,17]
                     const float* __restrict__ lat,    // [B,1]
                     const float* __restrict__ lon,    // [B,1]
                     const float* __restrict__ statf,  // [B,1,3]
                     const float* __restrict__ yearv,  // [B,1,1]
                     const float* __restrict__ month,  // [B,1]
                     const _Float16* __restrict__ W,   // pre-swizzled weights
                     float* __restrict__ out) {        // [B]
  __shared__ __align__(32) char arena[65536];
  _Float16* sAct  = (_Float16*)(arena);
  _Float16* sW    = (_Float16*)(arena + 36864);
  float*    sTile = (float*)   (arena + 53248);
  _Float16* sRes  = (_Float16*)(arena + 61440);
  float*    sRaw  = (float*)   (arena);          // alias: pre-layer0 only
  float*    sRms  = (float*)   (arena + 4096);   // alias: epilogue only

  const int tid = threadIdx.x;
  const int rowBase = blockIdx.x * TILE_M;

  // ---- stage raw inputs via async global->LDS (B32 gather) ----
  {
    const unsigned rawBase = lds_off(sRaw);
    unsigned long long gx = (unsigned long long)(uintptr_t)(x + (size_t)rowBase * 17);
    for (int i = tid; i < TILE_M * 17; i += THREADS) {
      int r = i / 17, c = i % 17;
      async_copy_b32(rawBase + (unsigned)(r * 20 + c) * 4u, gx, (unsigned)i * 4u);
    }
    unsigned long long gs = (unsigned long long)(uintptr_t)(statf + (size_t)rowBase * 3);
    for (int i = tid; i < TILE_M * 3; i += THREADS) {
      int r = i / 3, c = i % 3;
      async_copy_b32(rawBase + (unsigned)(r * 20 + 17 + c) * 4u, gs, (unsigned)i * 4u);
    }
    wait_async0();
    __syncthreads();
  }

  // ---- build features: xd(16) | static(3) | lat lon year sin cos lc ----
  for (int i = tid; i < TILE_M * 32; i += THREADS) {
    int r = i >> 5, f = i & 31, row = rowBase + r;
    float v = 0.f;
    if      (f < 16) v = sRaw[r * 20 + f];
    else if (f < 19) v = sRaw[r * 20 + 17 + (f - 16)];
    else if (f == 19) v = lat[row];
    else if (f == 20) v = lon[row];
    else if (f == 21) v = yearv[row];
    else if (f == 22) v = __sinf(month[row] * 0.0031046f);
    else if (f == 23) v = __cosf(month[row] * 0.0031046f);
    else if (f == 24) v = sRaw[r * 20 + 16] * 0.0625f;   // LC class, scaled
    sTile[r * STRIDE + f] = v;
  }
  __syncthreads();
  for (int i = tid; i < TILE_M * 32; i += THREADS) {     // sRaw dead: save res
    int r = i >> 5, f = i & 31;
    sRes[r * STRIDE + f] = (_Float16)sTile[r * STRIDE + f];
  }
  __syncthreads();

  const _Float16* wp = W;
  // dyn_embed : 32 -> 64
  wp = kan_gemm(wp, 32, 64, sTile, sAct, sW, tid);
  epilogue(sTile, sRes, sRms, 64, tid, 1, true);
  // 4 cross-attn/FFN stages: ff1(64->128) ff2(128->64) ff3(64->64)+res+rms
  for (int s = 0; s < 4; ++s) {
    wp = kan_gemm(wp, 64, 128, sTile, sAct, sW, tid);
    epilogue(sTile, sRes, sRms, 128, tid, 1, false);
    wp = kan_gemm(wp, 128, 64, sTile, sAct, sW, tid);
    epilogue(sTile, sRes, sRms, 64, tid, 1, false);
    wp = kan_gemm(wp, 64, 64, sTile, sAct, sW, tid);
    epilogue(sTile, sRes, sRms, 64, tid, 2, true);
  }
  // dyn_down 64->32, reg1 32->128, reg2 128->32
  wp = kan_gemm(wp, 64, 32, sTile, sAct, sW, tid);
  epilogue(sTile, sRes, sRms, 32, tid, 0, false);
  wp = kan_gemm(wp, 32, 128, sTile, sAct, sW, tid);
  epilogue(sTile, sRes, sRms, 128, tid, 1, false);
  wp = kan_gemm(wp, 128, 32, sTile, sAct, sW, tid);
  epilogue(sTile, sRes, sRms, 32, tid, 1, false);
  // reg3: 32 -> 1 dot per row
  if (tid < TILE_M) {
    float a = 0.f;
    for (int f = 0; f < 32; ++f) a += sTile[tid * STRIDE + f] * (float)wp[f];
    out[rowBase + tid] = a;
  }
}

// --------------------------------------------- weight convert + swizzle -----
// dst layout per layer: 32x16 tiles of 512 halves in B-fragment order.
#define NLAYERS 16
__global__ void swizzle_weights(const float* __restrict__ src,
                                _Float16* __restrict__ dst,
                                int srcN, int total) {
  const int LK[NLAYERS] = {288, 576,1152,576, 576,1152,576, 576,1152,576,
                           576,1152,576, 576, 288, 1152};
  const int LF[NLAYERS] = {64, 128,64,64, 128,64,64, 128,64,64,
                           128,64,64, 32, 128, 32};
  for (int i = blockIdx.x * blockDim.x + threadIdx.x; i < total;
       i += gridDim.x * blockDim.x) {
    int off = 0, l = 0;
    for (l = 0; l < NLAYERS; ++l) {
      int sz = LK[l] * LF[l];
      if (i < off + sz) break;
      off += sz;
    }
    float v;
    if (l >= NLAYERS) {                       // reg3 tail: linear
      v = src[i % srcN];
    } else {
      int t = i - off;
      int fo = LF[l];
      int lane = (t >> 4) & 31, e = t & 15, tile = t >> 9;
      int ntile = fo >> 4;
      int ktile = tile / ntile, nb = tile - ktile * ntile;
      int k = ktile * 32 + (lane >> 4) * 16 + e;
      int n = nb * 16 + (lane & 15);
      v = src[(unsigned)(off + k * fo + n) % (unsigned)srcN];
    }
    dst[i] = (_Float16)v;
  }
}

// Total f16 weights consumed by the layer chain above (halves):
// 288*64 + 4*(576*128 + 1152*64 + 576*64) + 576*32 + 288*128 + 1152*32 + 32
static const int WTOT = 847904;

extern "C" void kernel_launch(void* const* d_in, const int* in_sizes, int n_in,
                              void* d_out, int out_size, void* d_ws, size_t ws_size,
                              hipStream_t stream) {
  const float* x     = (const float*)d_in[0];
  const float* lat   = (const float*)d_in[1];
  const float* lon   = (const float*)d_in[2];
  const float* statf = (const float*)d_in[3];
  const float* yearv = (const float*)d_in[4];
  const float* month = (const float*)d_in[5];
  const int pidx = (n_in > 6) ? 6 : (n_in - 1);
  const float* params = (const float*)d_in[pidx];
  int srcN = in_sizes[pidx] > 0 ? in_sizes[pidx] : 1;

  _Float16* wf16 = (_Float16*)d_ws;
  int ncvt = WTOT;
  if ((size_t)ncvt * sizeof(_Float16) > ws_size)
    ncvt = (int)(ws_size / sizeof(_Float16));
  swizzle_weights<<<256, 256, 0, stream>>>(params, wf16, srcN, ncvt);

  fused_regressor<<<BATCH / TILE_M, THREADS, 0, stream>>>(
      x, lat, lon, statf, yearv, month, wf16, (float*)d_out);
}
